// DiffiTAttention_66468913873453
// MI455X (gfx1250) — matrix operand order repaired and checked by reading, not compiled, not run on hardware
//
#include <hip/hip_runtime.h>
#include <hip/hip_bf16.h>

// ---------------------------------------------------------------------------
// DiffiT windowed attention for gfx1250 (MI455X), wave32 + WMMA f16->f32.
// Pipeline: cvt -> bias gather -> tmod GEMM -> qkv GEMM -> flash-attn -> proj.
// CDNA5 paths: v_wmma_f32_16x16x32_f16, global_load_async_to_lds_b128 (+
// s_wait_asynccnt), global_prefetch_b8.
// ---------------------------------------------------------------------------

typedef _Float16 v16h __attribute__((ext_vector_type(16)));
typedef _Float16 v8h  __attribute__((ext_vector_type(8)));
typedef float    v8f  __attribute__((ext_vector_type(8)));

#define WMMA_F16F32(a, b, c) \
  __builtin_amdgcn_wmma_f32_16x16x32_f16(false, (a), false, (b), (short)0, (c), false, false)

static __device__ __forceinline__ v16h combine16(v8h lo, v8h hi) {
  return __builtin_shufflevector(lo, hi, 0, 1, 2, 3, 4, 5, 6, 7, 8, 9, 10, 11, 12, 13, 14, 15);
}

// Generic pointer -> 32-bit LDS byte offset.
static __device__ __forceinline__ unsigned lds_addr32(const void* p) {
  return (unsigned)(uintptr_t)(const __attribute__((address_space(3))) void*)p;
}

// CDNA5 async memory->LDS copy of 16 bytes (tracked by ASYNCcnt).
static __device__ __forceinline__ void async_g2l_b128(unsigned lds, const void* gaddr) {
  asm volatile("global_load_async_to_lds_b128 %0, %1, off"
               :: "v"(lds), "v"((unsigned long long)(uintptr_t)gaddr)
               : "memory");
}

static __device__ __forceinline__ void wait_asynccnt0() {
  asm volatile("s_wait_asynccnt 0x0" ::: "memory");
}

// A fragment: 16x32 f16 tile of row-major A (leading dim lda).
// A layout: lane (&15) = row, 16 contiguous halves = K from (lane>>4)*16.
static __device__ __forceinline__ v16h load_a_frag(const _Float16* __restrict__ A, int lda,
                                                   int m0, int kb, int lane) {
  int row = m0 + (lane & 15);
  int k0  = kb + ((lane >> 4) << 4);
  const _Float16* p = A + (size_t)row * lda + k0;
  v8h lo = *(const v8h*)p;
  v8h hi = *(const v8h*)(p + 8);
  return combine16(lo, hi);
}

// B fragment: 32(K)x16(N) f16 tile. Bt is B transposed, row-major [N][K].
// B layout (same k order as A): lane (&15) = column, 16 contiguous halves = K from (lane>>4)*16.
static __device__ __forceinline__ v16h load_b_frag(const _Float16* __restrict__ Bt, int ldb,
                                                   int n0, int kb, int lane) {
  int col = n0 + (lane & 15);
  int k0  = kb + ((lane >> 4) << 4);
  const _Float16* p = Bt + (size_t)col * ldb + k0;
  v8h lo = *(const v8h*)p;
  v8h hi = *(const v8h*)(p + 8);
  return combine16(lo, hi);
}

// One wave computes a 32x32 C tile (2x2 WMMA tiles) of A[MxK] @ Bt[NxK]^T.
static __device__ __forceinline__ void gemm_core(const _Float16* __restrict__ A,
                                                 const _Float16* __restrict__ Bt,
                                                 int K, int m0, int n0, int lane,
                                                 v8f c[2][2]) {
  for (int kb = 0; kb < K; kb += 32) {
    v16h a0 = load_a_frag(A, K, m0,      kb, lane);
    v16h a1 = load_a_frag(A, K, m0 + 16, kb, lane);
    v16h b0 = load_b_frag(Bt, K, n0,      kb, lane);
    v16h b1 = load_b_frag(Bt, K, n0 + 16, kb, lane);
    c[0][0] = WMMA_F16F32(a0, b0, c[0][0]);
    c[0][1] = WMMA_F16F32(a0, b1, c[0][1]);
    c[1][0] = WMMA_F16F32(a1, b0, c[1][0]);
    c[1][1] = WMMA_F16F32(a1, b1, c[1][1]);
  }
}

// ---------------------------- conversion kernels ---------------------------

__global__ void diffit_cvt_f16_kernel(const float* __restrict__ in, _Float16* __restrict__ out,
                                      int n) {
  int i = blockIdx.x * blockDim.x + threadIdx.x;
  if (i < n) out[i] = (_Float16)in[i];
}

// in: row-major [K][N] f32  ->  out: row-major [N][K] f16 (the "Bt" layout)
__global__ void diffit_cvtT_kernel(const float* __restrict__ in, _Float16* __restrict__ out,
                                   int K, int N) {
  int i = blockIdx.x * blockDim.x + threadIdx.x;
  if (i < K * N) {
    int n = i / K, k = i - n * K;
    out[i] = (_Float16)in[(size_t)k * N + n];
  }
}

// biasg[h][q*256+k] = rpb_table[rp_index[q][k]][h], f16
__global__ void diffit_bias_kernel(const int* __restrict__ rp, const float* __restrict__ tab,
                                   _Float16* __restrict__ biasg) {
  int i = blockIdx.x * blockDim.x + threadIdx.x;  // [0, 16*65536)
  int h = i >> 16, e = i & 65535;
  biasg[i] = (_Float16)tab[rp[e] * 16 + h];
}

// ------------------------------ GEMM kernels -------------------------------

// tmod[b][j] = temb[b] . temb_w[:,j] + temb_b[j] + qkv_b[j]   (M=256,K=512,N=1536)
__global__ __launch_bounds__(256) void diffit_tmod_kernel(
    const _Float16* __restrict__ tembh, const _Float16* __restrict__ tembwT,
    const float* __restrict__ temb_b, const float* __restrict__ qkv_b,
    _Float16* __restrict__ tmodh) {
  int lane = threadIdx.x & 31, w = threadIdx.x >> 5;
  int m0 = blockIdx.y * 64 + (w & 1) * 32;
  int n0 = blockIdx.x * 128 + (w >> 1) * 32;
  v8f c[2][2] = {};
  gemm_core(tembh, tembwT, 512, m0, n0, lane, c);
  int hf = lane >> 4, lx = lane & 15;
#pragma unroll
  for (int rt = 0; rt < 2; ++rt)
#pragma unroll
    for (int ct = 0; ct < 2; ++ct)
#pragma unroll
      for (int r = 0; r < 8; ++r) {
        int row = m0 + rt * 16 + r + hf * 8;
        int col = n0 + ct * 16 + lx;
        tmodh[(size_t)row * 1536 + col] = (_Float16)(c[rt][ct][r] + temb_b[col] + qkv_b[col]);
      }
}

// qkv = x @ qkv_w + tmod[b]; scatter into q (row-major), k (row-major == Bt of
// K^T), vT ([d][n] == Bt of V).  M=65536, K=512, N=1536.
__global__ __launch_bounds__(256) void diffit_qkv_kernel(
    const _Float16* __restrict__ xh, const _Float16* __restrict__ qkvwT,
    const _Float16* __restrict__ tmodh,
    _Float16* __restrict__ qh, _Float16* __restrict__ kh, _Float16* __restrict__ vth) {
  int lane = threadIdx.x & 31, w = threadIdx.x >> 5;
  int m0 = blockIdx.y * 64 + (w & 1) * 32;
  int n0 = blockIdx.x * 128 + (w >> 1) * 32;
  v8f c[2][2] = {};
  gemm_core(xh, qkvwT, 512, m0, n0, lane, c);
  int hf = lane >> 4, lx = lane & 15;
#pragma unroll
  for (int rt = 0; rt < 2; ++rt)
#pragma unroll
    for (int ct = 0; ct < 2; ++ct)
#pragma unroll
      for (int r = 0; r < 8; ++r) {
        int row = m0 + rt * 16 + r + hf * 8;   // b*256 + n
        int col = n0 + ct * 16 + lx;           // [0,1536)
        int bb = row >> 8, n = row & 255;
        float val = c[rt][ct][r] + (float)tmodh[(size_t)bb * 1536 + col];
        int which = col >> 9, cc = col & 511, hh = cc >> 5, d = cc & 31;
        size_t bh = (size_t)bb * 16 + hh;
        _Float16 hv = (_Float16)val;
        if (which == 0)      qh[(bh * 256 + n) * 32 + d] = hv;
        else if (which == 1) kh[(bh * 256 + n) * 32 + d] = hv;
        else                 vth[(bh * 32 + d) * 256 + n] = hv;
      }
}

// ---------------------------- attention kernel -----------------------------
// One workgroup per (b,h): flash-attention over 256 keys in chunks of 32.
__global__ __launch_bounds__(256) void diffit_attn_kernel(
    const _Float16* __restrict__ qh, const _Float16* __restrict__ kh,
    const _Float16* __restrict__ vth, const _Float16* __restrict__ biasg,
    _Float16* __restrict__ abuf) {
  const int bh = blockIdx.x;
  const int b = bh >> 4, h = bh & 15;
  const _Float16* q    = qh  + (size_t)bh * (256 * 32);
  const _Float16* kk   = kh  + (size_t)bh * (256 * 32);
  const _Float16* vt   = vth + (size_t)bh * (32 * 256);
  const _Float16* bias = biasg + (size_t)h * (256 * 256);

  __shared__ _Float16 sk[256 * 32];   // K, row-major [key][d]  (Bt of K^T)
  __shared__ _Float16 sv[32 * 256];   // V^T, row-major [d][key] (Bt of V)
  __shared__ _Float16 sp[8][32 * 40]; // per-wave P scratch, padded stride 40

  const int t = threadIdx.x;
  // Stage K and V^T into LDS with CDNA5 async memory->LDS copies (ASYNCcnt).
  {
    const unsigned lk = lds_addr32(&sk[0]);
    const unsigned lv = lds_addr32(&sv[0]);
#pragma unroll
    for (int j = 0; j < 4; ++j) {
      int off = t * 32 + j * 8;              // halves; 16 bytes per copy
      async_g2l_b128(lk + off * 2, kk + off);
      async_g2l_b128(lv + off * 2, vt + off);
    }
  }
  wait_asynccnt0();
  __syncthreads();

  const int wave = t >> 5, lane = t & 31;
  const int hf = lane >> 4, lx = lane & 15;
  const int qbase = wave * 32;

  v16h aq[2];
  aq[0] = load_a_frag(q, 32, qbase,      0, lane);
  aq[1] = load_a_frag(q, 32, qbase + 16, 0, lane);

  float m_[2][8], l_[2][8];
  v8f o[2][2] = {};
#pragma unroll
  for (int rt = 0; rt < 2; ++rt)
#pragma unroll
    for (int r = 0; r < 8; ++r) { m_[rt][r] = -1e30f; l_[rt][r] = 0.0f; }

  const float scale = 0.17677669529663687f;  // 1/sqrt(32)

  for (int kb = 0; kb < 256; kb += 32) {
    // Prefetch next chunk's bias rows into cache while we compute this chunk.
    if (kb + 32 < 256)
      __builtin_prefetch(&bias[(qbase + hf * 8) * 256 + kb + 32], 0, 3);

    // K fragments from LDS (Bt layout: contiguous d per key)
    v16h bk[2];
#pragma unroll
    for (int ct = 0; ct < 2; ++ct) {
      int key = kb + ct * 16 + lx;
      const _Float16* p = &sk[key * 32 + hf * 16];
      bk[ct] = combine16(*(const v8h*)p, *(const v8h*)(p + 8));
    }
    // S = q @ k^T  (hd = 32 == one WMMA K step)
    v8f s[2][2];
#pragma unroll
    for (int rt = 0; rt < 2; ++rt)
#pragma unroll
      for (int ct = 0; ct < 2; ++ct) {
        v8f z = {};
        s[rt][ct] = WMMA_F16F32(aq[rt], bk[ct], z);
      }
    // scale + relative-position bias
#pragma unroll
    for (int rt = 0; rt < 2; ++rt)
#pragma unroll
      for (int ct = 0; ct < 2; ++ct)
#pragma unroll
        for (int r = 0; r < 8; ++r) {
          int row = qbase + rt * 16 + r + hf * 8;
          int key = kb + ct * 16 + lx;
          s[rt][ct][r] = s[rt][ct][r] * scale + (float)bias[row * 256 + key];
        }
    // online softmax per row-tile
#pragma unroll
    for (int rt = 0; rt < 2; ++rt) {
      float mx[8];
#pragma unroll
      for (int r = 0; r < 8; ++r) mx[r] = fmaxf(s[rt][0][r], s[rt][1][r]);
#pragma unroll
      for (int msk = 1; msk < 16; msk <<= 1)
#pragma unroll
        for (int r = 0; r < 8; ++r) mx[r] = fmaxf(mx[r], __shfl_xor(mx[r], msk, 32));
      float alpha[8], rs[8];
#pragma unroll
      for (int r = 0; r < 8; ++r) {
        float mn = fmaxf(m_[rt][r], mx[r]);
        alpha[r] = __expf(m_[rt][r] - mn);
        m_[rt][r] = mn;
        float e0 = __expf(s[rt][0][r] - mn);
        float e1 = __expf(s[rt][1][r] - mn);
        s[rt][0][r] = e0; s[rt][1][r] = e1;
        rs[r] = e0 + e1;
      }
#pragma unroll
      for (int msk = 1; msk < 16; msk <<= 1)
#pragma unroll
        for (int r = 0; r < 8; ++r) rs[r] += __shfl_xor(rs[r], msk, 32);
#pragma unroll
      for (int r = 0; r < 8; ++r) l_[rt][r] = l_[rt][r] * alpha[r] + rs[r];
#pragma unroll
      for (int dt = 0; dt < 2; ++dt)
#pragma unroll
        for (int r = 0; r < 8; ++r) o[rt][dt][r] *= alpha[r];
      // spill P tile (C layout) to per-wave LDS scratch as f16
#pragma unroll
      for (int ct = 0; ct < 2; ++ct)
#pragma unroll
        for (int r = 0; r < 8; ++r)
          sp[wave][(rt * 16 + r + hf * 8) * 40 + ct * 16 + lx] = (_Float16)s[rt][ct][r];
    }
    // reload P in A layout (same-wave DS ops are in-order; no barrier needed)
    v16h ap[2];
#pragma unroll
    for (int rt = 0; rt < 2; ++rt) {
      const _Float16* p = &sp[wave][(rt * 16 + lx) * 40 + hf * 16];
      ap[rt] = combine16(*(const v8h*)p, *(const v8h*)(p + 8));
    }
    // V fragments from LDS (Bt layout via V^T)
    v16h bv[2];
#pragma unroll
    for (int dt = 0; dt < 2; ++dt) {
      int d = dt * 16 + lx;
      const _Float16* p = &sv[d * 256 + kb + hf * 16];
      bv[dt] = combine16(*(const v8h*)p, *(const v8h*)(p + 8));
    }
    // O += P @ V
#pragma unroll
    for (int rt = 0; rt < 2; ++rt)
#pragma unroll
      for (int dt = 0; dt < 2; ++dt)
        o[rt][dt] = WMMA_F16F32(ap[rt], bv[dt], o[rt][dt]);
  }

  // normalize and write [B,N,C] f16 (head-interleaved)
#pragma unroll
  for (int rt = 0; rt < 2; ++rt)
#pragma unroll
    for (int dt = 0; dt < 2; ++dt)
#pragma unroll
      for (int r = 0; r < 8; ++r) {
        int row = qbase + rt * 16 + r + hf * 8;
        int col = h * 32 + dt * 16 + lx;
        float val = o[rt][dt][r] / l_[rt][r];
        abuf[((size_t)b * 256 + row) * 512 + col] = (_Float16)val;
      }
}

// out = abuf @ proj_w + proj_b  (M=65536, K=512, N=512), f32 output
__global__ __launch_bounds__(256) void diffit_proj_kernel(
    const _Float16* __restrict__ ah, const _Float16* __restrict__ projwT,
    const float* __restrict__ proj_b, float* __restrict__ out) {
  int lane = threadIdx.x & 31, w = threadIdx.x >> 5;
  int m0 = blockIdx.y * 64 + (w & 1) * 32;
  int n0 = blockIdx.x * 128 + (w >> 1) * 32;
  v8f c[2][2] = {};
  gemm_core(ah, projwT, 512, m0, n0, lane, c);
  int hf = lane >> 4, lx = lane & 15;
#pragma unroll
  for (int rt = 0; rt < 2; ++rt)
#pragma unroll
    for (int ct = 0; ct < 2; ++ct)
#pragma unroll
      for (int r = 0; r < 8; ++r) {
        int row = m0 + rt * 16 + r + hf * 8;
        int col = n0 + ct * 16 + lx;
        out[(size_t)row * 512 + col] = c[rt][ct][r] + proj_b[col];
      }
}

// ------------------------------- launcher ----------------------------------

extern "C" void kernel_launch(void* const* d_in, const int* in_sizes, int n_in,
                              void* d_out, int out_size, void* d_ws, size_t ws_size,
                              hipStream_t stream) {
  (void)in_sizes; (void)n_in; (void)out_size; (void)ws_size;
  const float* x        = (const float*)d_in[0];
  const float* temb     = (const float*)d_in[1];
  const float* qkv_w    = (const float*)d_in[2];
  const float* qkv_b    = (const float*)d_in[3];
  const float* temb_w   = (const float*)d_in[4];
  const float* temb_b   = (const float*)d_in[5];
  const float* rpb      = (const float*)d_in[6];
  const float* proj_w   = (const float*)d_in[7];
  const float* proj_b   = (const float*)d_in[8];
  const int*   rp_index = (const int*)d_in[9];
  float* out = (float*)d_out;

  char* ws = (char*)d_ws;
  size_t off = 0;
  auto alloc_h = [&](size_t nhalf) -> _Float16* {
    off = (off + 255) & ~(size_t)255;
    _Float16* p = (_Float16*)(ws + off);
    off += nhalf * sizeof(_Float16);
    return p;
  };

  const size_t NX   = (size_t)65536 * 512;       // x / attention-output elems
  _Float16* xh     = alloc_h(NX);                // also reused as abuf (stream-ordered)
  _Float16* tembh  = alloc_h((size_t)256 * 512);
  _Float16* tmodh  = alloc_h((size_t)256 * 1536);
  _Float16* qkvwT  = alloc_h((size_t)1536 * 512);
  _Float16* tembwT = alloc_h((size_t)1536 * 512);
  _Float16* projwT = alloc_h((size_t)512 * 512);
  _Float16* biasg  = alloc_h((size_t)16 * 65536);
  _Float16* qh     = alloc_h((size_t)4096 * 256 * 32);
  _Float16* kh     = alloc_h((size_t)4096 * 256 * 32);
  _Float16* vth    = alloc_h((size_t)4096 * 32 * 256);
  _Float16* abuf   = xh;  // alias: x consumed before attention writes

  // Stage 0: conversions + bias gather
  diffit_cvt_f16_kernel<<<(int)(NX / 256), 256, 0, stream>>>(x, xh, (int)NX);
  diffit_cvt_f16_kernel<<<512, 256, 0, stream>>>(temb, tembh, 256 * 512);
  diffit_cvtT_kernel<<<3072, 256, 0, stream>>>(qkv_w, qkvwT, 512, 1536);
  diffit_cvtT_kernel<<<3072, 256, 0, stream>>>(temb_w, tembwT, 512, 1536);
  diffit_cvtT_kernel<<<1024, 256, 0, stream>>>(proj_w, projwT, 512, 512);
  diffit_bias_kernel<<<4096, 256, 0, stream>>>(rp_index, rpb, biasg);

  // Stage 1: temb modulation (fused temb_b + qkv_b)
  diffit_tmod_kernel<<<dim3(12, 4), 256, 0, stream>>>(tembh, tembwT, temb_b, qkv_b, tmodh);

  // Stage 2: QKV GEMM + scatter into q / k / v^T
  diffit_qkv_kernel<<<dim3(12, 1024), 256, 0, stream>>>(xh, qkvwT, tmodh, qh, kh, vth);

  // Stage 3: flash attention per (b, h)
  diffit_attn_kernel<<<4096, 256, 0, stream>>>(qh, kh, vth, biasg, abuf);

  // Stage 4: output projection
  diffit_proj_kernel<<<dim3(4, 1024), 256, 0, stream>>>(abuf, projwT, proj_b, out);
}
